// GATRegression_31078383354433
// MI455X (gfx1250) — matrix operand order repaired and checked by reading, not compiled, not run on hardware
//
#include <hip/hip_runtime.h>
#include <stdint.h>

// ---------------------------------------------------------------------------
// GAT 2-layer regression for MI455X (gfx1250, wave32).
//   prep: x,W1 -> bf16 (W1 transposed)            (streaming, trivial)
//   L1:   h1 = x@W1 via v_wmma_f32_16x16x32_bf16  (async-LDS staged A strip)
//   edge softmax (4 heads) -> wave-per-edge aggregate (L2-resident atomics)
//   ELU -> GEMV (W2) -> scalar edge softmax -> aggregate -> +b2
// ---------------------------------------------------------------------------

typedef __attribute__((ext_vector_type(16))) __bf16 v16bf;
typedef __attribute__((ext_vector_type(8)))  float  v8f;
typedef unsigned short u16;
typedef unsigned int   u32;

#define IN_C   128
#define HIDC   256   // HEADS*HID
#define HEADS  4
#define HID    64
#define NEG_SLOPE 0.2f

__device__ __forceinline__ u16 bfbits(float f) {
  u32 u = __float_as_uint(f);
  return (u16)((u + 0x7FFFu + ((u >> 16) & 1u)) >> 16);   // round-to-nearest-even
}

struct U8x32 { uint4 lo, hi; };
__device__ __forceinline__ v16bf as_v16bf(uint4 lo, uint4 hi) {
  U8x32 t{lo, hi};
  return __builtin_bit_cast(v16bf, t);
}

__device__ __forceinline__ float wred(float v) {
  #pragma unroll
  for (int o = 16; o; o >>= 1) v += __shfl_xor(v, o, 32);
  return v;
}

__device__ __forceinline__ void atomicMaxF32(float* addr, float v) {
  if (v >= 0.0f) atomicMax((int*)addr, __float_as_int(v));
  else           atomicMin((unsigned int*)addr, __float_as_uint(v));
}

__device__ __forceinline__ float lrelu(float x) {
  return x > 0.0f ? x : NEG_SLOPE * x;
}

// ---------------------------------------------------------------------------
// K0: init scratch (agg1=0, m=-inf, denoms=0, agg2=0)
__global__ __launch_bounds__(256) void k_init(float* agg1, float* m1, float* den1,
                                              float* m2, float* den2, float* agg2,
                                              int Nn) {
  long long i = (long long)blockIdx.x * 256 + threadIdx.x;
  long long tot = (long long)Nn * HIDC;
  if (i < tot) agg1[i] = 0.0f;
  if (i < (long long)Nn * HEADS) { m1[i] = -__builtin_inff(); den1[i] = 0.0f; }
  if (i < Nn) { m2[i] = -__builtin_inff(); den2[i] = 0.0f; agg2[i] = 0.0f; }
}

// ---------------------------------------------------------------------------
// Kp1: x (f32 row-major) -> xb (bf16 row-major), packed 2 per dword
__global__ __launch_bounds__(256) void k_prep_x(const float* __restrict__ x,
                                                u32* __restrict__ xb, long long npairs) {
  long long i = (long long)blockIdx.x * 256 + threadIdx.x;
  if (i >= npairs) return;
  float2 p = *(const float2*)(x + 2 * i);
  xb[i] = (u32)bfbits(p.x) | ((u32)bfbits(p.y) << 16);
}

// Kp2: W1 [128x256 f32 row-major] -> W1t [256x128 bf16 col-major (transposed)]
__global__ __launch_bounds__(256) void k_prep_w(const float* __restrict__ W1,
                                                u32* __restrict__ W1t) {
  int i = blockIdx.x * 256 + threadIdx.x;            // 256*64 pair slots
  if (i >= HIDC * (IN_C / 2)) return;
  int col = i >> 6;                                  // 0..255
  int k   = (i & 63) * 2;                            // 0..126
  float f0 = W1[(size_t)k * HIDC + col];
  float f1 = W1[(size_t)(k + 1) * HIDC + col];
  W1t[(size_t)col * (IN_C / 2) + (i & 63)] = (u32)bfbits(f0) | ((u32)bfbits(f1) << 16);
}

// ---------------------------------------------------------------------------
// K1: h1 = x @ W1 via v_wmma_f32_16x16x32_bf16.
// One block = one 16-row M-strip; A strip (4KB bf16) staged to LDS with
// global_load_async_to_lds_b128; 8 waves x 2 N-tiles each = 256 cols.
__global__ __launch_bounds__(256) void k_gemm1(const u16* __restrict__ xb,
                                               const u16* __restrict__ W1t,
                                               float* __restrict__ h1, int Nn) {
  __shared__ __align__(16) u16 Abuf[16 * IN_C];      // 4 KB

  const int tid  = threadIdx.x;
  const int row0 = blockIdx.x * 16;

  // ---- stage A strip into LDS: 256 threads x 16 bytes, async DMA ----
  {
    const int r  = tid >> 4;            // 0..15 row
    const int kc = (tid & 15) * 8;      // 0..120 (8 bf16 = 16 B)
    unsigned long long ga =
        (unsigned long long)(uintptr_t)(xb + (size_t)(row0 + r) * IN_C + kc);
    unsigned lo = (unsigned)(uintptr_t)(&Abuf[r * IN_C + kc]);
    asm volatile("global_load_async_to_lds_b128 %0, %1, off"
                 :: "v"(lo), "v"(ga) : "memory");
  }
  asm volatile("s_wait_asynccnt 0x0" ::: "memory");
  __syncthreads();

  const int lane = tid & 31;
  const int wave = tid >> 5;
  const int half = lane >> 4, lrow = lane & 15;

  const int colA = (2 * wave) * 16 + lrow;       // tile 0 column
  const int colB = (2 * wave + 1) * 16 + lrow;   // tile 1 column
  const u16* bcolA = W1t + (size_t)colA * IN_C;
  const u16* bcolB = W1t + (size_t)colB * IN_C;

  v8f c0 = {}, c1 = {};
  #pragma unroll
  for (int kb = 0; kb < 4; ++kb) {
    const int k0 = kb * 32;
    // A 16x32 bf16: lanes 0-15 K=k0..k0+15, lanes 16-31 K=k0+16..k0+31 (matches B)
    const int ka0 = k0 + half * 16;
    const int ka1 = k0 + half * 16 + 8;
    uint4 alo = *(const uint4*)(&Abuf[lrow * IN_C + ka0]);  // ds_load_b128
    uint4 ahi = *(const uint4*)(&Abuf[lrow * IN_C + ka1]);
    v16bf a = as_v16bf(alo, ahi);
    // B 32x16 bf16: lanes 0-15 K=k0..k0+15, lanes 16-31 K=k0+16..k0+31 (contig in W1t)
    const int kbb = k0 + half * 16;
    uint4 b0lo = *(const uint4*)(bcolA + kbb);
    uint4 b0hi = *(const uint4*)(bcolA + kbb + 8);
    uint4 b1lo = *(const uint4*)(bcolB + kbb);
    uint4 b1hi = *(const uint4*)(bcolB + kbb + 8);
    c0 = __builtin_amdgcn_wmma_f32_16x16x32_bf16(false, a, false, as_v16bf(b0lo, b0hi),
                                                 (short)0, c0, false, false);
    c1 = __builtin_amdgcn_wmma_f32_16x16x32_bf16(false, a, false, as_v16bf(b1lo, b1hi),
                                                 (short)0, c1, false, false);
  }
  // C/D: VGPR r -> M = r (lanes 0-15) / r+8 (lanes 16-31), N = lane&15
  const int rowb = row0 + half * 8;
  #pragma unroll
  for (int r = 0; r < 8; ++r) {
    h1[(size_t)(rowb + r) * HIDC + colA] = c0[r];
    h1[(size_t)(rowb + r) * HIDC + colB] = c1[r];
  }
}

// ---------------------------------------------------------------------------
// K2: per-node attention scores for layer 1 (wave per node)
__global__ __launch_bounds__(256) void k_scores1(const float* __restrict__ h1,
                                                 const float* __restrict__ att_src,
                                                 const float* __restrict__ att_dst,
                                                 float* __restrict__ ssrc,
                                                 float* __restrict__ sdst, int Nn) {
  const int node = blockIdx.x * 8 + (threadIdx.x >> 5);
  const int lane = threadIdx.x & 31;
  if (node >= Nn) return;
  float as_[HEADS] = {0.f, 0.f, 0.f, 0.f};
  float ad_[HEADS] = {0.f, 0.f, 0.f, 0.f};
  const float* row = h1 + (size_t)node * HIDC;
  #pragma unroll
  for (int j = 0; j < 8; ++j) {
    const int cidx = lane + 32 * j;
    const int h = cidx >> 6, f = cidx & 63;
    const float v = row[cidx];
    as_[h] += v * att_src[h * HID + f];
    ad_[h] += v * att_dst[h * HID + f];
  }
  #pragma unroll
  for (int h = 0; h < HEADS; ++h) {
    float a = wred(as_[h]);
    float d = wred(ad_[h]);
    if (lane == 0) { ssrc[node * HEADS + h] = a; sdst[node * HEADS + h] = d; }
  }
}

// ---------------------------------------------------------------------------
// edge fetch (with appended self loops)
__device__ __forceinline__ void edge_sd(const long long* ei, int e, int E,
                                        int& s, int& d) {
  if (e < E) { s = (int)ei[e]; d = (int)ei[E + e]; }
  else       { s = d = e - E; }
}

// K3: segment max (layer 1)
__global__ __launch_bounds__(256) void k_emax1(const long long* __restrict__ ei,
                                               const float* __restrict__ ssrc,
                                               const float* __restrict__ sdst,
                                               float* __restrict__ m1, int E, int ET) {
  const int e = blockIdx.x * 256 + threadIdx.x;
  if (e >= ET) return;
  int s, d; edge_sd(ei, e, E, s, d);
  #pragma unroll
  for (int h = 0; h < HEADS; ++h) {
    float v = lrelu(ssrc[s * HEADS + h] + sdst[d * HEADS + h]);
    atomicMaxF32(&m1[d * HEADS + h], v);
  }
}

// K4: exp + denominator (layer 1)
__global__ __launch_bounds__(256) void k_esum1(const long long* __restrict__ ei,
                                               const float* __restrict__ ssrc,
                                               const float* __restrict__ sdst,
                                               const float* __restrict__ m1,
                                               float* __restrict__ den1, int E, int ET) {
  const int e = blockIdx.x * 256 + threadIdx.x;
  if (e >= ET) return;
  int s, d; edge_sd(ei, e, E, s, d);
  #pragma unroll
  for (int h = 0; h < HEADS; ++h) {
    float v = lrelu(ssrc[s * HEADS + h] + sdst[d * HEADS + h]);
    float mv = m1[d * HEADS + h];
    if (mv < -1e30f) mv = 0.0f;
    atomicAdd(&den1[d * HEADS + h], __expf(v - mv));
  }
}

// K5: weighted aggregation (wave per edge, 256 feats, atomics into L2-resident agg1)
__global__ __launch_bounds__(256) void k_agg1(const long long* __restrict__ ei,
                                              const float* __restrict__ ssrc,
                                              const float* __restrict__ sdst,
                                              const float* __restrict__ m1,
                                              const float* __restrict__ den1,
                                              const float* __restrict__ h1,
                                              float* __restrict__ agg1, int E, int ET) {
  const int e = blockIdx.x * 8 + (threadIdx.x >> 5);
  const int lane = threadIdx.x & 31;
  if (e >= ET) return;
  int s, d; edge_sd(ei, e, E, s, d);
  float alpha[HEADS];
  #pragma unroll
  for (int h = 0; h < HEADS; ++h) {
    float v = lrelu(ssrc[s * HEADS + h] + sdst[d * HEADS + h]);
    float mv = m1[d * HEADS + h];
    if (mv < -1e30f) mv = 0.0f;
    alpha[h] = __expf(v - mv) / den1[d * HEADS + h];
  }
  const float* hs = h1 + (size_t)s * HIDC;
  float* ad = agg1 + (size_t)d * HIDC;
  #pragma unroll
  for (int j = 0; j < 8; ++j) {
    const int cidx = lane + 32 * j;
    atomicAdd(&ad[cidx], alpha[j >> 1] * hs[cidx]);
  }
}

// ---------------------------------------------------------------------------
// K6: fused  u = elu(agg1+b1); h2 = u . W2;  layer-2 scores  (wave per node)
__global__ __launch_bounds__(256) void k_elu_gemv(const float* __restrict__ agg1,
                                                  const float* __restrict__ b1,
                                                  const float* __restrict__ W2,
                                                  const float* __restrict__ att_src2,
                                                  const float* __restrict__ att_dst2,
                                                  float* __restrict__ h2,
                                                  float* __restrict__ s2s,
                                                  float* __restrict__ s2d, int Nn) {
  const int node = blockIdx.x * 8 + (threadIdx.x >> 5);
  const int lane = threadIdx.x & 31;
  if (node >= Nn) return;
  const float* row = agg1 + (size_t)node * HIDC;
  float acc = 0.0f;
  #pragma unroll
  for (int j = 0; j < 8; ++j) {
    const int cidx = lane + 32 * j;
    float t = row[cidx] + b1[cidx];
    float u = t > 0.0f ? t : (__expf(t) - 1.0f);   // ELU
    acc += u * W2[cidx];
  }
  acc = wred(acc);
  if (lane == 0) {
    h2[node]  = acc;
    s2s[node] = acc * att_src2[0];
    s2d[node] = acc * att_dst2[0];
  }
}

// K7: segment max (layer 2, scalar)
__global__ __launch_bounds__(256) void k_emax2(const long long* __restrict__ ei,
                                               const float* __restrict__ s2s,
                                               const float* __restrict__ s2d,
                                               float* __restrict__ m2, int E, int ET) {
  const int e = blockIdx.x * 256 + threadIdx.x;
  if (e >= ET) return;
  int s, d; edge_sd(ei, e, E, s, d);
  atomicMaxF32(&m2[d], lrelu(s2s[s] + s2d[d]));
}

// K8: exp + denominator (layer 2)
__global__ __launch_bounds__(256) void k_esum2(const long long* __restrict__ ei,
                                               const float* __restrict__ s2s,
                                               const float* __restrict__ s2d,
                                               const float* __restrict__ m2,
                                               float* __restrict__ den2, int E, int ET) {
  const int e = blockIdx.x * 256 + threadIdx.x;
  if (e >= ET) return;
  int s, d; edge_sd(ei, e, E, s, d);
  float mv = m2[d]; if (mv < -1e30f) mv = 0.0f;
  atomicAdd(&den2[d], __expf(lrelu(s2s[s] + s2d[d]) - mv));
}

// K9: scalar aggregation (layer 2)
__global__ __launch_bounds__(256) void k_agg2(const long long* __restrict__ ei,
                                              const float* __restrict__ s2s,
                                              const float* __restrict__ s2d,
                                              const float* __restrict__ m2,
                                              const float* __restrict__ den2,
                                              const float* __restrict__ h2,
                                              float* __restrict__ agg2, int E, int ET) {
  const int e = blockIdx.x * 256 + threadIdx.x;
  if (e >= ET) return;
  int s, d; edge_sd(ei, e, E, s, d);
  float mv = m2[d]; if (mv < -1e30f) mv = 0.0f;
  float alpha = __expf(lrelu(s2s[s] + s2d[d]) - mv) / den2[d];
  atomicAdd(&agg2[d], alpha * h2[s]);
}

// K10: epilogue out = agg2 + b2
__global__ __launch_bounds__(256) void k_final(const float* __restrict__ agg2,
                                               const float* __restrict__ b2,
                                               float* __restrict__ out, int Nn) {
  const int n = blockIdx.x * 256 + threadIdx.x;
  if (n < Nn) out[n] = agg2[n] + b2[0];
}

// ---------------------------------------------------------------------------
extern "C" void kernel_launch(void* const* d_in, const int* in_sizes, int n_in,
                              void* d_out, int out_size, void* d_ws, size_t ws_size,
                              hipStream_t stream) {
  const float*     x        = (const float*)d_in[0];
  const long long* ei       = (const long long*)d_in[1];   // int64 edge_index [2,E]
  const float*     W1       = (const float*)d_in[2];
  const float*     att_src1 = (const float*)d_in[3];
  const float*     att_dst1 = (const float*)d_in[4];
  const float*     b1       = (const float*)d_in[5];
  const float*     W2       = (const float*)d_in[6];
  const float*     att_src2 = (const float*)d_in[7];
  const float*     att_dst2 = (const float*)d_in[8];
  const float*     b2       = (const float*)d_in[9];
  float*           out      = (float*)d_out;

  const int N  = in_sizes[0] / IN_C;
  const int E  = in_sizes[1] / 2;
  const int ET = E + N;           // + self loops

  // workspace carve-up
  float* ws    = (float*)d_ws;
  float* h1    = ws;                        // N*256 f32
  float* agg1  = h1    + (size_t)N * HIDC;  // N*256 f32
  float* ssrc1 = agg1  + (size_t)N * HIDC;  // N*4
  float* sdst1 = ssrc1 + (size_t)N * HEADS;
  float* m1    = sdst1 + (size_t)N * HEADS;
  float* den1  = m1    + (size_t)N * HEADS;
  float* h2    = den1  + (size_t)N * HEADS; // N
  float* s2s   = h2   + N;
  float* s2d   = s2s  + N;
  float* m2    = s2d  + N;
  float* den2  = m2   + N;
  float* agg2  = den2 + N;
  u16*   xb    = (u16*)(agg2 + N);          // N*128 bf16
  u16*   W1t   = xb + (size_t)N * IN_C;     // 256*128 bf16 (transposed)

  const int TB = 256;
  const long long initN = (long long)N * HIDC;
  k_init<<<(unsigned)((initN + TB - 1) / TB), TB, 0, stream>>>(agg1, m1, den1, m2, den2, agg2, N);

  const long long xpairs = (long long)N * (IN_C / 2);
  k_prep_x<<<(unsigned)((xpairs + TB - 1) / TB), TB, 0, stream>>>(x, (u32*)xb, xpairs);
  k_prep_w<<<(HIDC * (IN_C / 2) + TB - 1) / TB, TB, 0, stream>>>(W1, (u32*)W1t);

  k_gemm1<<<N / 16, TB, 0, stream>>>(xb, W1t, h1, N);     // 50000/16 = 3125 strips

  k_scores1<<<(N + 7) / 8, TB, 0, stream>>>(h1, att_src1, att_dst1, ssrc1, sdst1, N);

  k_emax1<<<(ET + TB - 1) / TB, TB, 0, stream>>>(ei, ssrc1, sdst1, m1, E, ET);
  k_esum1<<<(ET + TB - 1) / TB, TB, 0, stream>>>(ei, ssrc1, sdst1, m1, den1, E, ET);
  k_agg1 <<<(ET + 7) / 8,      TB, 0, stream>>>(ei, ssrc1, sdst1, m1, den1, h1, agg1, E, ET);

  k_elu_gemv<<<(N + 7) / 8, TB, 0, stream>>>(agg1, b1, W2, att_src2, att_dst2, h2, s2s, s2d, N);

  k_emax2<<<(ET + TB - 1) / TB, TB, 0, stream>>>(ei, s2s, s2d, m2, E, ET);
  k_esum2<<<(ET + TB - 1) / TB, TB, 0, stream>>>(ei, s2s, s2d, m2, den2, E, ET);
  k_agg2 <<<(ET + TB - 1) / TB, TB, 0, stream>>>(ei, s2s, s2d, m2, den2, h2, agg2, E, ET);

  k_final<<<(N + TB - 1) / TB, TB, 0, stream>>>(agg2, b2, out, N);
}
